// RNNClassification_35682588295712
// MI455X (gfx1250) — matrix operand order repaired and checked by reading, not compiled, not run on hardware
//
#include <hip/hip_runtime.h>
#include <math.h>

// ---------------------------------------------------------------------------
// Problem constants (from reference): T=512, B=128, D_IN=512, D_H=1024
// ---------------------------------------------------------------------------
#define T_STEPS 512
#define BATCH   128
#define D_IN    512
#define D_H     1024
#define RNN_BLOCKS 16   // persistent blocks for the recurrent kernel

typedef __bf16 bf16_t;
typedef __attribute__((ext_vector_type(16))) __bf16        v16bf;
typedef __attribute__((ext_vector_type(8)))  float         v8f;
typedef __attribute__((ext_vector_type(4)))  unsigned int  u32x4;
typedef __attribute__((ext_vector_type(4)))  int           i32x4;
typedef __attribute__((ext_vector_type(8)))  int           i32x8;

union FragBits { v16bf v; u32x4 q[2]; };

#if defined(__has_builtin)
#  if __has_builtin(__builtin_amdgcn_tensor_load_to_lds)
#    define USE_TDM 1
#  endif
#endif

// ---------------------------------------------------------------------------
// WMMA fragment loaders (bf16, wave32, 16x16x32).
// A (16xK row-major): lanes 0-15 hold M=lane, K chunks {0..7, 16..23};
//                     lanes 16-31 hold M=lane-16, K chunks {8..15, 24..31}.
// B from W^T[N][K] row-major: lanes 0-15 hold N=lane, K chunks {0..7, 16..23};
//                     lanes 16-31 hold N=lane-16, K chunks {8..15, 24..31}.
// ---------------------------------------------------------------------------
__device__ __forceinline__ v16bf load_frag_a(const bf16_t* __restrict__ src,
                                             int ld, int lane) {
  const int m    = lane & 15;
  const int half = lane >> 4;
  const bf16_t* p = src + (size_t)m * ld + half * 8;
  FragBits f;
  f.q[0] = *(const u32x4*)(p);
  f.q[1] = *(const u32x4*)(p + 16);
  return f.v;
}

__device__ __forceinline__ v16bf load_frag_b(const bf16_t* __restrict__ srcT,
                                             int ld, int lane) {
  const int n    = lane & 15;
  const int half = lane >> 4;
  const bf16_t* p = srcT + (size_t)n * ld + half * 8;
  FragBits f;
  f.q[0] = *(const u32x4*)(p);
  f.q[1] = *(const u32x4*)(p + 16);
  return f.v;
}

// ---------------------------------------------------------------------------
// Conversion / packing kernels
// ---------------------------------------------------------------------------
__global__ void cvt_f32_to_bf16(const float* __restrict__ in,
                                bf16_t* __restrict__ out, int n) {
  int i = blockIdx.x * blockDim.x + threadIdx.x;
  if (i < n) out[i] = (bf16_t)in[i];
}

// in[rows][cols] f32  ->  out[cols][rows] bf16  (W^T pack for B operand)
__global__ void transpose_cvt_bf16(const float* __restrict__ in,
                                   bf16_t* __restrict__ out,
                                   int rows, int cols) {
  int i = blockIdx.x * blockDim.x + threadIdx.x;
  if (i < rows * cols) {
    int r = i / cols, c = i - r * cols;
    out[(size_t)c * rows + r] = (bf16_t)in[i];
  }
}

// ---------------------------------------------------------------------------
// Encoder GEMM: Wx[t,b,h] = X @ W_enc + b_enc
// M = T*B = 65536, K = 512, N = 1024. One wave computes a 16x64 tile.
// ---------------------------------------------------------------------------
__global__ void enc_gemm_kernel(const bf16_t* __restrict__ Xb,
                                const bf16_t* __restrict__ WencT,
                                const float* __restrict__ b_enc,
                                float* __restrict__ Wx) {
  const int lane = threadIdx.x & 31;
  const int wave = threadIdx.x >> 5;
  const int gw   = blockIdx.x * 8 + wave;    // 65536 waves total
  const int m0   = (gw >> 4) << 4;           // M tile base (16-row blocks)
  const int n0   = (gw & 15) << 6;           // N tile base (64-col blocks)

  v8f acc[4];
#pragma unroll
  for (int q = 0; q < 4; ++q) {
    float bv = b_enc[n0 + q * 16 + (lane & 15)];
    acc[q] = (v8f){bv, bv, bv, bv, bv, bv, bv, bv};
  }

  for (int k = 0; k < D_IN; k += 32) {
    v16bf a = load_frag_a(Xb + (size_t)m0 * D_IN + k, D_IN, lane);
#pragma unroll
    for (int q = 0; q < 4; ++q) {
      v16bf b = load_frag_b(WencT + (size_t)(n0 + q * 16) * D_IN + k, D_IN, lane);
      acc[q] = __builtin_amdgcn_wmma_f32_16x16x32_bf16(
          false, a, false, b, (short)0, acc[q], false, false);
    }
  }

  const int mrow = (lane >> 4) << 3;   // 0 or 8
  const int nc   = lane & 15;
#pragma unroll
  for (int q = 0; q < 4; ++q)
#pragma unroll
    for (int r = 0; r < 8; ++r)
      Wx[(size_t)(m0 + mrow + r) * D_H + n0 + q * 16 + nc] = acc[q][r];
}

// ---------------------------------------------------------------------------
// TDM: 1-D tile load (nelems bf16 elements, contiguous) global -> LDS.
// clang-23 / amdgpu-toolchain 6-arg builtin form; descriptor per
// CDNA5 ISA (D# group0 + group1; groups 2/3 disabled).
// ---------------------------------------------------------------------------
#ifdef USE_TDM
__device__ __forceinline__ void tdm_load_1d(unsigned lds_off,
                                            const void* gsrc,
                                            unsigned nelems) {
  unsigned long long ga = (unsigned long long)(uintptr_t)gsrc;
  u32x4 g0 = { 1u,                                   // count=1 (valid D#)
               lds_off,                               // lds_addr
               (unsigned)ga,                          // global_addr[31:0]
               (unsigned)((ga >> 32) & 0x01FFFFFFu) | (2u << 30) }; // hi + type=2
  i32x8 g1 = { (int)0x00010000,                      // data_size=1 (2B), mask=0
               (int)((nelems & 0xFFFFu) << 16),      // tensor_dim0 lo16
               (int)(((nelems >> 16) & 0xFFFFu) | (1u << 16)), // dim0 hi | dim1=1
               (int)((nelems & 0xFFFFu) << 16),      // tile_dim0
               1,                                     // tile_dim1=1, tile_dim2=0
               (int)nelems,                           // tensor_dim0_stride lo32
               0, 0 };
  i32x4 z4 = {0, 0, 0, 0};
  i32x8 z8 = {0, 0, 0, 0, 0, 0, 0, 0};
  __builtin_amdgcn_tensor_load_to_lds(g0, g1, z4, z4, z8, 0);
}
#endif

// ---------------------------------------------------------------------------
// Grid-wide barrier (monotonic counter, reset by memset each launch)
// ---------------------------------------------------------------------------
__device__ __forceinline__ void grid_sync(unsigned* bar, unsigned target) {
  __threadfence();
  __syncthreads();
  if (threadIdx.x == 0) {
    __hip_atomic_fetch_add(bar, 1u, __ATOMIC_RELEASE, __HIP_MEMORY_SCOPE_AGENT);
    while (__hip_atomic_load(bar, __ATOMIC_ACQUIRE, __HIP_MEMORY_SCOPE_AGENT) < target)
      __builtin_amdgcn_s_sleep(2);
  }
  __syncthreads();
}

// ---------------------------------------------------------------------------
// Persistent recurrent kernel: h_t = tanh(Wx_t + h_{t-1} @ W_h + b_h)
// 16 blocks x 8 waves; block (b&7) owns M rows [16*(b&7), +16),
// N half (b>>3)*512; each wave owns 64 N columns. Per step: TDM pulls the
// 16 needed h_prev rows (32 KB bf16) into LDS, then 32 k-steps x 4 WMMAs.
// ---------------------------------------------------------------------------
__global__ void rnn_scan_kernel(const float* __restrict__ Wx,
                                const bf16_t* __restrict__ WhT,
                                const float* __restrict__ b_h,
                                bf16_t* __restrict__ hbuf0,
                                bf16_t* __restrict__ hbuf1,
                                float* __restrict__ out,
                                unsigned* __restrict__ bar) {
  __shared__ __align__(16) bf16_t shA[16 * D_H];   // 32 KB

  const int lane = threadIdx.x & 31;
  const int wave = threadIdx.x >> 5;                        // 0..7
  const int m0   = (blockIdx.x & 7) << 4;                   // row block
  const int n0   = ((blockIdx.x >> 3) << 9) + (wave << 6);  // col block (64)
  const int mrow = (lane >> 4) << 3;
  const int nc   = lane & 15;

  float bias[4];
#pragma unroll
  for (int q = 0; q < 4; ++q) bias[q] = b_h[n0 + q * 16 + nc];

  bf16_t* hb[2] = {hbuf0, hbuf1};

  for (int t = 0; t < T_STEPS; ++t) {
    const bf16_t* hprev = hb[t & 1];
    bf16_t*       hnext = hb[(t + 1) & 1];

    // Stage h_prev rows [m0, m0+16) x D_H into LDS.
#ifdef USE_TDM
    if (wave == 0) {
      tdm_load_1d((unsigned)(size_t)(void*)shA,
                  hprev + (size_t)m0 * D_H, 16u * D_H);
      __builtin_amdgcn_s_wait_tensorcnt((unsigned short)0);
    }
#else
    {
      const u32x4* src = (const u32x4*)(hprev + (size_t)m0 * D_H);
      u32x4* dst = (u32x4*)shA;
      for (int i = threadIdx.x; i < (16 * D_H * 2) / 16; i += 256)
        dst[i] = src[i];
    }
#endif
    __syncthreads();

    // Accumulators initialized with the precomputed input projection Wx_t.
    const float* wxt = Wx + ((size_t)t * BATCH + m0) * D_H + n0;
    v8f acc[4];
#pragma unroll
    for (int q = 0; q < 4; ++q)
#pragma unroll
      for (int r = 0; r < 8; ++r)
        acc[q][r] = wxt[(size_t)(mrow + r) * D_H + q * 16 + nc];

    if (t + 1 < T_STEPS)   // prewarm next step's Wx tile in L2
      __builtin_prefetch(wxt + (size_t)BATCH * D_H, 0, 1);

    for (int k = 0; k < D_H; k += 32) {
      // A fragment from LDS (h_prev rows), B fragments streamed from L2.
      FragBits fa;
      {
        const bf16_t* p = shA + (size_t)(lane & 15) * D_H + k + ((lane >> 4) << 3);
        fa.q[0] = *(const u32x4*)(p);
        fa.q[1] = *(const u32x4*)(p + 16);
      }
#pragma unroll
      for (int q = 0; q < 4; ++q) {
        v16bf b = load_frag_b(WhT + (size_t)(n0 + q * 16) * D_H + k, D_H, lane);
        acc[q] = __builtin_amdgcn_wmma_f32_16x16x32_bf16(
            false, fa.v, false, b, (short)0, acc[q], false, false);
      }
    }

    // tanh + bias; last step writes f32 result, otherwise bf16 state.
#pragma unroll
    for (int q = 0; q < 4; ++q)
#pragma unroll
      for (int r = 0; r < 8; ++r) {
        float val = tanhf(acc[q][r] + bias[q]);
        size_t idx = (size_t)(m0 + mrow + r) * D_H + n0 + q * 16 + nc;
        if (t == T_STEPS - 1) out[idx] = val;
        else                  hnext[idx] = (bf16_t)val;
      }

    grid_sync(bar, (unsigned)(t + 1) * RNN_BLOCKS);
  }
}

// ---------------------------------------------------------------------------
// Host launch
// ---------------------------------------------------------------------------
extern "C" void kernel_launch(void* const* d_in, const int* in_sizes, int n_in,
                              void* d_out, int out_size, void* d_ws, size_t ws_size,
                              hipStream_t stream) {
  const float* X     = (const float*)d_in[0];
  const float* h0    = (const float*)d_in[1];
  const float* W_enc = (const float*)d_in[2];
  const float* b_enc = (const float*)d_in[3];
  const float* W_h   = (const float*)d_in[4];
  const float* b_h   = (const float*)d_in[5];
  float* out = (float*)d_out;

  // Workspace layout (bytes)
  char* ws = (char*)d_ws;
  size_t off = 0;
  float*  Wx    = (float*)(ws + off);  off += (size_t)T_STEPS * BATCH * D_H * 4;  // 268 MB
  bf16_t* Xb    = (bf16_t*)(ws + off); off += (size_t)T_STEPS * BATCH * D_IN * 2; // 67 MB
  bf16_t* WencT = (bf16_t*)(ws + off); off += (size_t)D_IN * D_H * 2;             // 1 MB
  bf16_t* WhT   = (bf16_t*)(ws + off); off += (size_t)D_H * D_H * 2;              // 2 MB
  bf16_t* hbuf0 = (bf16_t*)(ws + off); off += (size_t)BATCH * D_H * 2;
  bf16_t* hbuf1 = (bf16_t*)(ws + off); off += (size_t)BATCH * D_H * 2;
  unsigned* bar = (unsigned*)(ws + off);

  (void)hipMemsetAsync(bar, 0, 64, stream);

  const int nx = T_STEPS * BATCH * D_IN;        // 33,554,432
  cvt_f32_to_bf16<<<nx / 256, 256, 0, stream>>>(X, Xb, nx);
  transpose_cvt_bf16<<<(D_IN * D_H) / 256, 256, 0, stream>>>(W_enc, WencT, D_IN, D_H);
  transpose_cvt_bf16<<<(D_H * D_H) / 256, 256, 0, stream>>>(W_h, WhT, D_H, D_H);
  cvt_f32_to_bf16<<<(BATCH * D_H) / 256, 256, 0, stream>>>(h0, hbuf0, BATCH * D_H);

  // 65536 waves / 8 per block = 8192 blocks
  enc_gemm_kernel<<<8192, 256, 0, stream>>>(Xb, WencT, b_enc, Wx);

  rnn_scan_kernel<<<RNN_BLOCKS, 256, 0, stream>>>(Wx, WhT, b_h, hbuf0, hbuf1, out, bar);
}
